// SelfAttentionHead_78769700209202
// MI455X (gfx1250) — matrix operand order repaired and checked by reading, not compiled, not run on hardware
//
#include <hip/hip_runtime.h>

typedef __attribute__((ext_vector_type(16))) _Float16 v16h;
typedef __attribute__((ext_vector_type(2)))  _Float16 h2;
typedef __attribute__((ext_vector_type(8)))  float    v8f;

#define T_SEQ   4096
#define BATCH   8
#define C_EMB   1024
#define C_HEAD  64
#define MTOT    (BATCH * T_SEQ)   // 32768
#define XS_LD   34

__device__ __forceinline__ v8f wmma_f16(v16h a, v16h b, v8f c) {
  return __builtin_amdgcn_wmma_f32_16x16x32_f16(false, a, false, b, (short)0, c, false, false);
}

// A-frag (16x32 f16, M x K): lane holds row M = lane&15; pairs contiguous in memory (row-major, K contiguous)
__device__ __forceinline__ v16h load_a_frag(const _Float16* __restrict__ base, int ld, int lane) {
  v16h a;
  int M = lane & 15, half = lane >> 4;
#pragma unroll
  for (int v = 0; v < 8; ++v) {
    int kb = ((v >> 2) * 16) + (half * 8) + ((v & 3) * 2);
    h2 p = *(const h2*)(base + (long)M * ld + kb);
    a[2 * v]     = p.x;
    a[2 * v + 1] = p.y;
  }
  return a;
}

// B-frag (32x16 f16, K x N) where memory holds B[k][n] = mem[n*ld + k] (pairs contiguous along k)
__device__ __forceinline__ v16h load_b_frag_t(const _Float16* __restrict__ mem, long ld, int lane) {
  v16h b;
  int N = lane & 15, half = lane >> 4;
#pragma unroll
  for (int v = 0; v < 8; ++v) {
    h2 p = *(const h2*)(mem + (long)N * ld + ((v >> 2) * 16) + (half * 8) + ((v & 3) * 2));
    b[2 * v]     = p.x;
    b[2 * v + 1] = p.y;
  }
  return b;
}

// ---------------- Kernel 0: W -> Wt f16 transpose ([192][1024]) -----------
__global__ __launch_bounds__(256) void wprep_kernel(
    const float* __restrict__ Wq, const float* __restrict__ Wk,
    const float* __restrict__ Wv, _Float16* __restrict__ Wt)
{
  int blk = blockIdx.x;             // 48 blocks
  int w = blk >> 4;                 // 0=Q,1=K,2=V
  int kbase = (blk & 15) * 64;
  const float* W = (w == 0) ? Wq : ((w == 1) ? Wk : Wv);
  for (int idx = threadIdx.x; idx < 64 * 64; idx += 256) {
    int kl = idx >> 6, n = idx & 63;      // coalesced read along n
    Wt[((long)(w * 64 + n)) * C_EMB + kbase + kl] =
        (_Float16)W[(long)(kbase + kl) * C_HEAD + n];
  }
}

// ---------------- Kernel 1: fused QKV projection --------------------------
__global__ __launch_bounds__(256) void qkv_proj_kernel(
    const float* __restrict__ x, const _Float16* __restrict__ Wt,
    _Float16* __restrict__ Qf, _Float16* __restrict__ Kf, _Float16* __restrict__ Vt)
{
  __shared__ _Float16 xs[128 * XS_LD];
  const int tid  = threadIdx.x;
  const int lane = tid & 31;
  const int wv   = tid >> 5;
  const long row0 = (long)blockIdx.x * 128;
  const int N = lane & 15, half = lane >> 4;

  v8f acc[12];
  v8f zero = {0.f, 0.f, 0.f, 0.f, 0.f, 0.f, 0.f, 0.f};
#pragma unroll
  for (int i = 0; i < 12; ++i) acc[i] = zero;

  for (int kc = 0; kc < C_EMB / 32; ++kc) {
    // stage 128x32 x-tile (f32 -> f16), coalesced b128 loads
#pragma unroll
    for (int i = 0; i < 4; ++i) {
      int e4 = tid + 256 * i;           // 0..1023 float4s
      int r  = e4 >> 3;                 // 8 float4 per 32-col row
      int c4 = e4 & 7;
      float4 vx = *(const float4*)(x + (row0 + r) * (long)C_EMB + kc * 32 + c4 * 4);
      _Float16* dst = &xs[r * XS_LD + c4 * 4];
      dst[0] = (_Float16)vx.x; dst[1] = (_Float16)vx.y;
      dst[2] = (_Float16)vx.z; dst[3] = (_Float16)vx.w;
    }
    __syncthreads();

    // A-frag for this wave's 16 rows (ds_load_2addr_b32 pairs)
    v16h a;
#pragma unroll
    for (int v = 0; v < 8; ++v) {
      int kb = ((v >> 2) * 16) + (half * 8) + ((v & 3) * 2);
      h2 p = *(const h2*)&xs[(wv * 16 + (lane & 15)) * XS_LD + kb];
      a[2 * v] = p.x; a[2 * v + 1] = p.y;
    }
    // B-frags straight from L2-resident transposed weights (contiguous pairs)
#pragma unroll
    for (int nt = 0; nt < 12; ++nt) {
      v16h b = load_b_frag_t(Wt + (long)(nt * 16) * C_EMB + kc * 32, C_EMB, lane);
      acc[nt] = wmma_f16(a, b, acc[nt]);
    }
    __syncthreads();
  }

  // store results: Q,K row-major f16; V transposed [64][MTOT] f16
#pragma unroll
  for (int nt = 0; nt < 12; ++nt) {
    int w   = nt >> 2;                 // 0=Q, 1=K, 2=V (matches Wt row order)
    int col = (nt * 16 + N) & 63;
#pragma unroll
    for (int j = 0; j < 8; ++j) {
      long row = row0 + wv * 16 + j + 8 * half;
      _Float16 hv = (_Float16)acc[nt][j];
      if (w == 0)      Qf[row * C_HEAD + col] = hv;
      else if (w == 1) Kf[row * C_HEAD + col] = hv;
      else             Vt[(long)col * MTOT + row] = hv;
    }
  }
}

// ---------------- Kernel 2: per-key-column softmax stats ------------------
__global__ __launch_bounds__(32) void colstats_kernel(
    const _Float16* __restrict__ Qf, const _Float16* __restrict__ Kf,
    float* __restrict__ mOut, float* __restrict__ dOut)
{
  const int lane = threadIdx.x;
  const int b  = blockIdx.x >> 8;
  const int kt = blockIdx.x & 255;
  const int k0 = kt * 16;
  const _Float16* Qb = Qf + (long)b * T_SEQ * C_HEAD;
  const _Float16* Kb = Kf + (long)b * T_SEQ * C_HEAD;
  const int N = lane & 15, half = lane >> 4;
  const float scale = 0.125f;   // 1/sqrt(64)
  v8f zero = {0.f, 0.f, 0.f, 0.f, 0.f, 0.f, 0.f, 0.f};

  // K-tile B-frags (keys = columns), constant over q loop
  v16h bk0 = load_b_frag_t(Kb + (long)k0 * C_HEAD,      C_HEAD, lane);
  v16h bk1 = load_b_frag_t(Kb + (long)k0 * C_HEAD + 32, C_HEAD, lane);

  float rm = -1e30f, rs = 0.f;
  for (int qt = kt; qt < T_SEQ / 16; ++qt) {
    int q0 = qt * 16;
    v16h a0 = load_a_frag(Qb + (long)q0 * C_HEAD,      C_HEAD, lane);
    v16h a1 = load_a_frag(Qb + (long)q0 * C_HEAD + 32, C_HEAD, lane);
    v8f sacc = zero;
    sacc = wmma_f16(a0, bk0, sacc);
    sacc = wmma_f16(a1, bk1, sacc);

    float s[8];
    float tmax = -1e30f;
#pragma unroll
    for (int j = 0; j < 8; ++j) {
      int q = q0 + j + 8 * half;
      float v = sacc[j] * scale;
      v = (q >= k0 + N) ? v : -1e30f;   // causal mask (branch-free)
      s[j] = v;
      tmax = fmaxf(tmax, v);
    }
    tmax = fmaxf(tmax, __shfl_xor(tmax, 16, 32));  // combine the two column halves
    float newm = fmaxf(rm, tmax);
    float corr = __expf(rm - newm);
    float psum = 0.f;
#pragma unroll
    for (int j = 0; j < 8; ++j) psum += __expf(s[j] - newm);
    rs = rs * corr + psum;
    rm = newm;
  }
  rs += __shfl_xor(rs, 16, 32);
  if (half == 0) {
    mOut[b * T_SEQ + k0 + N] = rm;
    dOut[b * T_SEQ + k0 + N] = rs;
  }
}

// ---------------- Kernel 3: output = P @ V --------------------------------
__global__ __launch_bounds__(32) void attnout_kernel(
    const _Float16* __restrict__ Qf, const _Float16* __restrict__ Kf,
    const _Float16* __restrict__ Vt, const float* __restrict__ mIn,
    const float* __restrict__ dIn, float* __restrict__ out)
{
  __shared__ _Float16 pS[16 * XS_LD];
  const int lane = threadIdx.x;
  const int b  = blockIdx.x >> 8;
  const int qt = blockIdx.x & 255;
  const int q0 = qt * 16;
  const _Float16* Qb = Qf + (long)b * T_SEQ * C_HEAD;
  const _Float16* Kb = Kf + (long)b * T_SEQ * C_HEAD;
  const int N = lane & 15, half = lane >> 4;
  const float scale = 0.125f;
  v8f zero = {0.f, 0.f, 0.f, 0.f, 0.f, 0.f, 0.f, 0.f};

  v16h aq0 = load_a_frag(Qb + (long)q0 * C_HEAD,      C_HEAD, lane);
  v16h aq1 = load_a_frag(Qb + (long)q0 * C_HEAD + 32, C_HEAD, lane);

  v8f oacc[4];
#pragma unroll
  for (int i = 0; i < 4; ++i) oacc[i] = zero;

  for (int kc0 = 0; kc0 <= q0; kc0 += 32) {
#pragma unroll
    for (int sub = 0; sub < 2; ++sub) {
      int k0 = kc0 + sub * 16;
      v16h bk0 = load_b_frag_t(Kb + (long)k0 * C_HEAD,      C_HEAD, lane);
      v16h bk1 = load_b_frag_t(Kb + (long)k0 * C_HEAD + 32, C_HEAD, lane);
      v8f sacc = zero;
      sacc = wmma_f16(aq0, bk0, sacc);
      sacc = wmma_f16(aq1, bk1, sacc);
      float mc   = mIn[b * T_SEQ + k0 + N];
      float invd = 1.f / dIn[b * T_SEQ + k0 + N];
#pragma unroll
      for (int j = 0; j < 8; ++j) {
        int q = q0 + j + 8 * half;
        float p = (q >= k0 + N) ? __expf(sacc[j] * scale - mc) * invd : 0.f;
        pS[(j + 8 * half) * XS_LD + sub * 16 + N] = (_Float16)p;
      }
    }
    __syncthreads();
    // reload P as A-frag (C-layout -> A-layout via LDS)
    v16h ap;
#pragma unroll
    for (int v = 0; v < 8; ++v) {
      int kb = ((v >> 2) * 16) + (half * 8) + ((v & 3) * 2);
      h2 p = *(const h2*)&pS[(lane & 15) * XS_LD + kb];
      ap[2 * v] = p.x; ap[2 * v + 1] = p.y;
    }
#pragma unroll
    for (int ht = 0; ht < 4; ++ht) {
      v16h bv = load_b_frag_t(Vt + (long)(ht * 16) * MTOT + (long)b * T_SEQ + kc0,
                              (long)MTOT, lane);
      oacc[ht] = wmma_f16(ap, bv, oacc[ht]);
    }
    __syncthreads();
  }

#pragma unroll
  for (int ht = 0; ht < 4; ++ht)
#pragma unroll
    for (int j = 0; j < 8; ++j) {
      long q = q0 + j + 8 * half;
      out[((long)b * T_SEQ + q) * C_HEAD + ht * 16 + N] = oacc[ht][j];
    }
}

extern "C" void kernel_launch(void* const* d_in, const int* in_sizes, int n_in,
                              void* d_out, int out_size, void* d_ws, size_t ws_size,
                              hipStream_t stream) {
  // setup_inputs order: x, Wk, Wq, Wv
  const float* x  = (const float*)d_in[0];
  const float* Wk = (const float*)d_in[1];
  const float* Wq = (const float*)d_in[2];
  const float* Wv = (const float*)d_in[3];
  float* out = (float*)d_out;

  char* ws = (char*)d_ws;
  _Float16* Qf = (_Float16*)(ws);                            // 4 MB   [MTOT][64] f16
  _Float16* Kf = (_Float16*)(ws + (size_t)4  * 1024 * 1024); // 4 MB   [MTOT][64] f16
  _Float16* Vt = (_Float16*)(ws + (size_t)8  * 1024 * 1024); // 4 MB   [64][MTOT] f16
  float*    mB = (float*)   (ws + (size_t)12 * 1024 * 1024); // 128 KB [B*T] f32
  float*    dB = (float*)   (ws + (size_t)12 * 1024 * 1024 + 128 * 1024);
  _Float16* Wt = (_Float16*)(ws + (size_t)12 * 1024 * 1024 + 256 * 1024); // 384 KB [192][1024]

  wprep_kernel<<<48, 256, 0, stream>>>(Wq, Wk, Wv, Wt);
  qkv_proj_kernel<<<MTOT / 128, 256, 0, stream>>>(x, Wt, Qf, Kf, Vt);
  colstats_kernel<<<BATCH * (T_SEQ / 16), 32, 0, stream>>>(Qf, Kf, mB, dB);
  attnout_kernel<<<BATCH * (T_SEQ / 16), 32, 0, stream>>>(Qf, Kf, Vt, mB, dB, out);
}
